// MessagePassing_52991306498426
// MI455X (gfx1250) — matrix operand order repaired and checked by reading, not compiled, not run on hardware
//
#include <hip/hip_runtime.h>

// ---------------------------------------------------------------------------
// MessagePassing fused GNN layer for gfx1250 (MI455X, wave32, WMMA)
//   msg  = node @ Wn                          (bf16 WMMA, f32 accum)
//   h    = LReLU(LReLU(edge@We1+be1)@We2+be2) (bf16 WMMA, f32 accum, fused)
//   out  = segment_sum(msg[idx_j] * h, seg_i) (gather + global_atomic_add_f32)
//
// Round-2 changes: fragment-major (pre-swizzled) weight storage in LDS so
// B-fragments load as 2x ds_load_b128 per lane; h1 A-fragments load as
// aligned v8bf (b128); conversions via __builtin_convertvector.
// ---------------------------------------------------------------------------

typedef __attribute__((ext_vector_type(16))) __bf16 v16bf;
typedef __attribute__((ext_vector_type(8)))  __bf16 v8bf;
typedef __attribute__((ext_vector_type(4)))  __bf16 v4bf;
typedef __attribute__((ext_vector_type(8)))  float  v8f;
typedef __attribute__((ext_vector_type(4)))  float  f32x4;

#define D_NODE 128
#define D_EDGE 32
#define D_HID  128
#define NEG_SLOPE 0.01f

static __device__ __forceinline__ float lrelu(float x) {
  return x >= 0.0f ? x : NEG_SLOPE * x;
}
static __device__ __forceinline__ v8bf join8(v4bf a, v4bf b) {
  return __builtin_shufflevector(a, b, 0, 1, 2, 3, 4, 5, 6, 7);
}
static __device__ __forceinline__ v16bf join16(v8bf a, v8bf b) {
  return __builtin_shufflevector(a, b, 0, 1, 2, 3, 4, 5, 6, 7,
                                       8, 9, 10, 11, 12, 13, 14, 15);
}

// --- A fragment (16x32 bf16) from global f32, ISA 7.12.2 layout -------------
// lane L holds row M=L&15; element e -> K = (e>>3)*16 + (L>>4)*8 + (e&7)
// => two contiguous 8-float runs at kbase+half*8 and kbase+16+half*8.
static __device__ __forceinline__ v16bf
load_a_frag_global(const float* __restrict__ base, int ld, int kbase, int lane) {
  int row = lane & 15, half = lane >> 4;
  const float* p = base + (size_t)row * ld + kbase + half * 8;
  f32x4 x0 = *(const f32x4*)(p);
  f32x4 x1 = *(const f32x4*)(p + 4);
  f32x4 x2 = *(const f32x4*)(p + 16);
  f32x4 x3 = *(const f32x4*)(p + 20);
  return join16(join8(__builtin_convertvector(x0, v4bf),
                      __builtin_convertvector(x1, v4bf)),
                join8(__builtin_convertvector(x2, v4bf),
                      __builtin_convertvector(x3, v4bf)));
}

// --- A fragment from row-major bf16 LDS tile (two b128 loads per lane) ------
static __device__ __forceinline__ v16bf
load_a_frag_lds(const __bf16* h, int ld, int kbase, int lane) {
  int row = lane & 15, half = lane >> 4;
  const __bf16* p = h + row * ld + kbase + half * 8;
  v8bf lo = *(const v8bf*)(p);        // K = kbase+half*8 .. +7
  v8bf hi = *(const v8bf*)(p + 16);   // K = kbase+16+half*8 .. +7
  return join16(lo, hi);
}

// --- B fragment from fragment-major (pre-swizzled) LDS weights --------------
// Storage: frag (kc,nt) occupies 32 lanes x 16 bf16 contiguously.
static __device__ __forceinline__ v16bf
load_b_frag_sw(const __bf16* w, int kc, int ntile, int lane) {
  const __bf16* p = w + (size_t)(((kc * 8 + ntile) * 32) + lane) * 16;
  v8bf lo = *(const v8bf*)(p);
  v8bf hi = *(const v8bf*)(p + 8);
  return join16(lo, hi);
}

// Pre-swizzle a K-major [Kdim][128] f32 weight into fragment-major bf16 LDS.
// B layout: lane L holds col N=L&15; element e -> K = (e>>3)*16 + (L>>4)*8 + (e&7) (per kc).
static __device__ __forceinline__ void
fill_b_swizzled(__bf16* dst, const float* __restrict__ W, int Kdim,
                int tid, int nthreads) {
  int total = Kdim * D_HID;
  for (int i = tid; i < total; i += nthreads) {
    int e    = i & 15;
    int lane = (i >> 4) & 31;
    int nt   = (i >> 9) & 7;
    int kc   = i >> 12;
    int k = kc * 32 + ((e >> 3) << 4) + ((lane >> 4) << 3) + (e & 7);
    int n = nt * 16 + (lane & 15);
    dst[i] = (__bf16)W[k * D_HID + n];
  }
}

// ---------------------------------------------------------------------------
__global__ void __launch_bounds__(256) zero_f32_kernel(float* __restrict__ p, int n) {
  for (int i = blockIdx.x * blockDim.x + threadIdx.x; i < n; i += gridDim.x * blockDim.x)
    p[i] = 0.0f;
}

// msg[t*16+m, :] = node[t*16+m, :] @ Wn   (one 16-row tile per wave)
__global__ void __launch_bounds__(256) node_proj_kernel(
    const float* __restrict__ node, const float* __restrict__ Wn,
    float* __restrict__ msg, int nTiles) {
  __shared__ __attribute__((aligned(16))) __bf16 sWn[D_NODE * D_HID];  // 32 KB
  fill_b_swizzled(sWn, Wn, D_NODE, threadIdx.x, blockDim.x);
  __syncthreads();

  int lane = threadIdx.x & 31;
  int wave = threadIdx.x >> 5;
  int wgid = blockIdx.x * (blockDim.x >> 5) + wave;
  int nWav = gridDim.x * (blockDim.x >> 5);
  int colb = lane & 15, half = lane >> 4;

  for (int t = wgid; t < nTiles; t += nWav) {
    v8f zero = {};
    v8f acc[8];
#pragma unroll
    for (int n = 0; n < 8; ++n) acc[n] = zero;
    const float* aBase = node + (size_t)t * 16 * D_NODE;
#pragma unroll
    for (int kc = 0; kc < 4; ++kc) {                        // K = 128 = 4x32
      v16bf a = load_a_frag_global(aBase, D_NODE, kc * 32, lane);
#pragma unroll
      for (int n = 0; n < 8; ++n) {                         // N = 128 = 8x16
        v16bf b = load_b_frag_sw(sWn, kc, n, lane);
        acc[n] = __builtin_amdgcn_wmma_f32_16x16x32_bf16(
            false, a, false, b, (short)0, acc[n], false, false);
      }
    }
    // C layout: lane holds (M = r + half*8, N = n*16 + colb)
#pragma unroll
    for (int n = 0; n < 8; ++n)
#pragma unroll
      for (int r = 0; r < 8; ++r) {
        int m = half * 8 + r;
        msg[((size_t)t * 16 + m) * D_HID + n * 16 + colb] = acc[n][r];
      }
  }
}

// Fused: edge MLP -> gather msg[idx_j] -> multiply -> atomic scatter to out
__global__ void __launch_bounds__(128) edge_fused_kernel(
    const float* __restrict__ edge, const float* __restrict__ We1,
    const float* __restrict__ be1,  const float* __restrict__ We2,
    const float* __restrict__ be2,  const int* __restrict__ seg_i,
    const int* __restrict__ idx_j,  const float* __restrict__ msg,
    float* __restrict__ out, int nTiles) {
  __shared__ __attribute__((aligned(16))) __bf16 sWe1[D_EDGE * D_HID];   // 8 KB
  __shared__ __attribute__((aligned(16))) __bf16 sWe2[D_HID * D_HID];    // 32 KB
  __shared__ float sBe1[D_HID];
  __shared__ float sBe2[D_HID];
  __shared__ __attribute__((aligned(16))) __bf16 sH1[4][16 * D_HID];     // 4x4 KB

  fill_b_swizzled(sWe1, We1, D_EDGE, threadIdx.x, blockDim.x);
  fill_b_swizzled(sWe2, We2, D_HID, threadIdx.x, blockDim.x);
  for (int i = threadIdx.x; i < D_HID; i += blockDim.x) {
    sBe1[i] = be1[i];
    sBe2[i] = be2[i];
  }
  __syncthreads();

  int lane = threadIdx.x & 31;
  int wave = threadIdx.x >> 5;
  __bf16* h1 = sH1[wave];                                  // wave-private tile
  int wgid = blockIdx.x * (blockDim.x >> 5) + wave;
  int nWav = gridDim.x * (blockDim.x >> 5);
  int colb = lane & 15, half = lane >> 4;

  for (int t = wgid; t < nTiles; t += nWav) {
    size_t e0 = (size_t)t * 16;
    if (t + nWav < nTiles)                                 // next tile's edges
      __builtin_prefetch(edge + (size_t)(t + nWav) * 16 * D_EDGE, 0, 1);

    v8f zero = {};
    v8f acc[8];
    // ---- layer 1: [16,32] @ [32,128], K = D_EDGE = 32 exactly -------------
    {
      v16bf a = load_a_frag_global(edge + e0 * D_EDGE, D_EDGE, 0, lane);
#pragma unroll
      for (int n = 0; n < 8; ++n) {
        v16bf b = load_b_frag_sw(sWe1, 0, n, lane);
        acc[n] = __builtin_amdgcn_wmma_f32_16x16x32_bf16(
            false, a, false, b, (short)0, zero, false, false);
      }
    }
    // bias + LeakyReLU, C-layout -> A-layout via wave-private LDS
#pragma unroll
    for (int n = 0; n < 8; ++n) {
      int col = n * 16 + colb;
      float bv = sBe1[col];
#pragma unroll
      for (int r = 0; r < 8; ++r) {
        int m = half * 8 + r;
        h1[m * D_HID + col] = (__bf16)lrelu(acc[n][r] + bv);
      }
    }
    __builtin_amdgcn_wave_barrier();  // per-wave LDS ops are in-order

    // ---- layer 2: [16,128] @ [128,128] ------------------------------------
#pragma unroll
    for (int n = 0; n < 8; ++n) acc[n] = zero;
#pragma unroll
    for (int kc = 0; kc < 4; ++kc) {
      v16bf a = load_a_frag_lds(h1, D_HID, kc * 32, lane);
#pragma unroll
      for (int n = 0; n < 8; ++n) {
        v16bf b = load_b_frag_sw(sWe2, kc, n, lane);
        acc[n] = __builtin_amdgcn_wmma_f32_16x16x32_bf16(
            false, a, false, b, (short)0, acc[n], false, false);
      }
    }
    // ---- bias + LeakyReLU + gather(msg) + scatter-add(out) ----------------
#pragma unroll
    for (int r = 0; r < 8; ++r) {
      int m = half * 8 + r;
      int j = idx_j[e0 + m];
      int s = seg_i[e0 + m];
      const float* mrow = msg + (size_t)j * D_HID;        // L2-resident (25.6MB)
      float* orow = out + (size_t)s * D_HID;              // L2-resident
#pragma unroll
      for (int n = 0; n < 8; ++n) {
        int col = n * 16 + colb;
        float h2 = lrelu(acc[n][r] + sBe2[col]);
        atomicAdd(&orow[col], h2 * mrow[col]);
      }
    }
  }
}

// ---------------------------------------------------------------------------
extern "C" void kernel_launch(void* const* d_in, const int* in_sizes, int n_in,
                              void* d_out, int out_size, void* d_ws, size_t ws_size,
                              hipStream_t stream) {
  const float* node = (const float*)d_in[0];
  const float* edge = (const float*)d_in[1];
  const float* Wn   = (const float*)d_in[2];
  const float* We1  = (const float*)d_in[3];
  const float* be1  = (const float*)d_in[4];
  const float* We2  = (const float*)d_in[5];
  const float* be2  = (const float*)d_in[6];
  const int* seg_i  = (const int*)d_in[7];
  const int* idx_j  = (const int*)d_in[8];
  float* out = (float*)d_out;
  float* msg = (float*)d_ws;     // needs nNodes * D_HID * 4 bytes (25.6 MB)

  int nNodes  = in_sizes[0] / D_NODE;
  int nEdges  = in_sizes[1] / D_EDGE;
  int nTilesN = nNodes / 16;     // 3125  (N divides 16 exactly)
  int nTilesE = nEdges / 16;     // 37500 (E divides 16 exactly)

  // 1) zero the accumulation target (harness poisons d_out)
  zero_f32_kernel<<<1024, 256, 0, stream>>>(out, out_size);

  // 2) msg = node @ Wn   (8 waves/block, 1 tile per wave per iteration)
  {
    int wpb = 8;
    int blocks = (nTilesN + wpb - 1) / wpb;
    if (blocks > 2048) blocks = 2048;
    node_proj_kernel<<<blocks, 256, 0, stream>>>(node, Wn, msg, nTilesN);
  }

  // 3) fused edge MLP + gather + scatter (4 waves/block, 57 KB LDS)
  {
    int blocks = 1024;
    edge_fused_kernel<<<blocks, 128, 0, stream>>>(
        edge, We1, be1, We2, be2, seg_i, idx_j, msg, out, nTilesE);
  }
}
